// MiMoV2Attention_2044404433736
// MI455X (gfx1250) — matrix operand order repaired and checked by reading, not compiled, not run on hardware
//
#include <hip/hip_runtime.h>
#include <hip/hip_bf16.h>
#include <math.h>

// ---------------------------------------------------------------------------
// MiMoV2 attention block for MI455X (gfx1250, wave32, WMMA).
//  * Big GEMMs: bf16x3 split-precision via V_WMMA_F32_16X16X32_BF16
//    (A=Ah+Al, B=Bh+Bl; C ~= AhBh + AhBl + AlBh; fp32 accumulate; ~1e-5 rel err)
//  * Attention core: exact fp32 via V_WMMA_F32_16X16X4_F32
//  * K/V tile staging: GLOBAL_LOAD_ASYNC_TO_LDS_B128 (ASYNCcnt) when available
// ---------------------------------------------------------------------------

typedef __attribute__((ext_vector_type(2)))  float  v2f;
typedef __attribute__((ext_vector_type(4)))  float  f32x4;
typedef __attribute__((ext_vector_type(8)))  float  v8f;
typedef __attribute__((ext_vector_type(4)))  __bf16 v4bf;
typedef __attribute__((ext_vector_type(8)))  __bf16 v8bf;
typedef __attribute__((ext_vector_type(16))) __bf16 v16bf;

#define HH 32          // query heads
#define KHH 8          // kv heads
#define GG 4           // group size
#define DD 128         // head dim
#define VDD 64         // v dim
#define QKVW 5632      // qkv row width  (32*128 + 8*128 + 8*64)
#define QOFF 4096      // k offset in qkv row
#define VOFF 5120      // v offset in qkv row
#define WIN 1024
#define HIDDEN_DIM 4096
#define OPROJ_K 2048   // H*VD
#define V_SCALE 1.25f

#if defined(__gfx1250__) &&                                          \
    __has_builtin(__builtin_amdgcn_global_load_async_to_lds_b128) && \
    __has_builtin(__builtin_amdgcn_s_wait_asynccnt)
#define USE_ASYNC_LDS 1
#else
#define USE_ASYNC_LDS 0
#endif

// exact parameter types per hipcc diagnostic:
//   __attribute__((__vector_size__(4*sizeof(int)))) int  AS(1)/AS(3) *
typedef int b128_t __attribute__((__vector_size__(4 * sizeof(int))));
typedef b128_t __attribute__((address_space(1)))* gb128_ptr;  // global src
typedef b128_t __attribute__((address_space(3)))* lb128_ptr;  // LDS dst

static __device__ inline gb128_ptr as_global_b128(const void* p) {
  // flat address of global memory == global address
  return (gb128_ptr)(unsigned long long)p;
}
static __device__ inline lb128_ptr as_lds_b128(const void* p) {
  // low 32 bits of a flat LDS-aperture address are the LDS offset
  return (lb128_ptr)(unsigned)(unsigned long long)p;
}

static __device__ inline v8f zero8() {
  v8f z;
#pragma unroll
  for (int i = 0; i < 8; ++i) z[i] = 0.0f;
  return z;
}

static __device__ inline v16bf join16(v8bf lo, v8bf hi) {
  return __builtin_shufflevector(lo, hi, 0, 1, 2, 3, 4, 5, 6, 7,
                                 8, 9, 10, 11, 12, 13, 14, 15);
}

// ---------------------------------------------------------------------------
// bf16x3 GEMM: C[M,N] = A[M,K] * B[K,N]  (fp32 in/out).
// Requires M%128==0, N%64==0, K%32==0.
// Block = 256 threads (8 waves); block tile 128x64; wave = 64x16 strip
// (4 WMMA m-subtiles); K chunk = 32 (one bf16 WMMA K-depth).
// Per chunk per wave: 4 subtiles x 3 split-products = 12 v_wmma bf16.
// ---------------------------------------------------------------------------
#define AKS 40   // LDS k-stride for A tiles (bf16) : 80B = 20 banks, 16B aligned
#define BKS 40   // LDS k-stride for B^T tiles (bf16)

__global__ __launch_bounds__(256) void gemm_bf16x3_kernel(
    const float* __restrict__ A, const float* __restrict__ B,
    float* __restrict__ C, int M, int N, int K) {
  __shared__ __align__(16) __bf16 Ah[128 * AKS];
  __shared__ __align__(16) __bf16 Al[128 * AKS];
  __shared__ __align__(16) __bf16 BTh[64 * BKS];  // transposed: [n][k]
  __shared__ __align__(16) __bf16 BTl[64 * BKS];

  const int tid  = threadIdx.x;
  const int lane = tid & 31;
  const int wv   = tid >> 5;
  const int hi   = lane >> 4;
  const int c    = lane & 15;
  const int wm   = wv >> 2;   // 0..1  (M 64-strip)
  const int wn   = wv & 3;    // 0..3  (N 16-strip)
  const int m_blk = blockIdx.x * 128;
  const int n_blk = blockIdx.y * 64;

  v8f acc[4];
#pragma unroll
  for (int i = 0; i < 4; ++i) acc[i] = zero8();

  // cooperative-load mapping
  const int la_m = tid >> 1;         // 0..127
  const int la_k = (tid & 1) * 16;   // 0 or 16
  const int lb_k = tid >> 3;         // 0..31
  const int lb_n = (tid & 7) * 8;    // 0..56

  for (int k0 = 0; k0 < K; k0 += 32) {
    // ---- A chunk 128x32: load fp32, split to (Ah, Al) bf16 in LDS ----
    const float* ag = A + (size_t)(m_blk + la_m) * K + k0 + la_k;
#pragma unroll
    for (int j = 0; j < 4; ++j) {
      f32x4 x = *(const f32x4*)(ag + 4 * j);
      v4bf h, l;
#pragma unroll
      for (int i = 0; i < 4; ++i) {
        h[i] = (__bf16)x[i];
        l[i] = (__bf16)(x[i] - (float)h[i]);
      }
      *(v4bf*)(&Ah[la_m * AKS + la_k + 4 * j]) = h;
      *(v4bf*)(&Al[la_m * AKS + la_k + 4 * j]) = l;
    }
    // ---- B chunk 32x64: load fp32, split + transpose into [n][k] LDS ----
    const float* bg = B + (size_t)(k0 + lb_k) * N + n_blk + lb_n;
#pragma unroll
    for (int j = 0; j < 8; ++j) {
      float x = bg[j];
      __bf16 h = (__bf16)x;
      BTh[(lb_n + j) * BKS + lb_k] = h;
      BTl[(lb_n + j) * BKS + lb_k] = (__bf16)(x - (float)h);
    }
    // prefetch next chunk into cache while we compute this one
    if (k0 + 32 < K) {
      __builtin_prefetch(ag + 32, 0, 3);
      __builtin_prefetch(bg + (size_t)32 * N, 0, 3);
    }
    __syncthreads();

    // B fragments: b[e] = B[k = e + 16*hi][n] -> contiguous 16 bf16 in B^T
    const int bbase = (wn * 16 + c) * BKS + 16 * hi;
    v16bf bh = join16(*(const v8bf*)(&BTh[bbase]), *(const v8bf*)(&BTh[bbase + 8]));
    v16bf bl = join16(*(const v8bf*)(&BTl[bbase]), *(const v8bf*)(&BTl[bbase + 8]));

#pragma unroll
    for (int mt = 0; mt < 4; ++mt) {
      // A fragments: a[e] = A[m][k = e + 16*hi]
      const int ab = (wm * 64 + mt * 16 + c) * AKS + 16 * hi;
      v16bf ah = join16(*(const v8bf*)(&Ah[ab]), *(const v8bf*)(&Ah[ab + 8]));
      v16bf al = join16(*(const v8bf*)(&Al[ab]), *(const v8bf*)(&Al[ab + 8]));
      acc[mt] = __builtin_amdgcn_wmma_f32_16x16x32_bf16(
          false, ah, false, bh, (short)0, acc[mt], false, false);
      acc[mt] = __builtin_amdgcn_wmma_f32_16x16x32_bf16(
          false, ah, false, bl, (short)0, acc[mt], false, false);
      acc[mt] = __builtin_amdgcn_wmma_f32_16x16x32_bf16(
          false, al, false, bh, (short)0, acc[mt], false, false);
    }
    __syncthreads();
  }

  // D layout: row = 8*hi + v, col = c
#pragma unroll
  for (int mt = 0; mt < 4; ++mt) {
    int mrow = m_blk + wm * 64 + mt * 16 + 8 * hi;
    float* cp = C + (size_t)mrow * N + n_blk + wn * 16 + c;
#pragma unroll
    for (int v = 0; v < 8; ++v) cp[(size_t)v * N] = acc[mt][v];
  }
}

// ---------------------------------------------------------------------------
// RoPE over Q (heads 0..31) and K (heads 32..39) of the qkv buffer, in place.
// grid = (T, H+KH), block = 64 (one thread per rotation pair).
// ---------------------------------------------------------------------------
__global__ __launch_bounds__(64) void rope_kernel(
    const int* __restrict__ positions, float* __restrict__ qkv) {
  const int t = blockIdx.x;
  const int head = blockIdx.y;
  const int d = threadIdx.x;  // 0..63
  float* p = (head < HH)
                 ? qkv + (size_t)t * QKVW + head * DD
                 : qkv + (size_t)t * QKVW + QOFF + (head - HH) * DD;
  const float pos = (float)positions[t];
  // inv_freq = THETA^(-d/64) = exp(-ln(1e6) * d / 64)
  const float inv = expf(-13.815510558f * ((float)d * (1.0f / 64.0f)));
  const float f = pos * inv;
  float s, cv;
  sincosf(f, &s, &cv);
  const float x1 = p[d];
  const float x2 = p[d + 64];
  p[d]      = x1 * cv - x2 * s;
  p[d + 64] = x2 * cv + x1 * s;
}

// ---------------------------------------------------------------------------
// Flash-style sliding-window attention with sink logits (exact fp32 WMMA).
// grid = (T/16, KH), block = 128 (4 waves = 4 GQA heads of one KV head).
// K/V tiles staged via async global->LDS (no VGPR round-trip) when available.
// V_SCALE is linear in V, so it is folded into the epilogue (acc/l * 1.25).
// ---------------------------------------------------------------------------
#define KST 132  // K tile LDS stride (floats)
#define VST 72   // V tile LDS stride
#define PST 20   // P staging stride

__global__ __launch_bounds__(128) void attn_kernel(
    const float* __restrict__ qkv, const float* __restrict__ sink,
    float* __restrict__ attn_out, int T) {
  __shared__ __align__(16) float Klds[16 * KST];
  __shared__ __align__(16) float Vlds[16 * VST];
  __shared__ float Plds[4 * 16 * PST];

  const int tid  = threadIdx.x;
  const int lane = tid & 31;
  const int wv   = tid >> 5;  // group head g
  const int hi   = lane >> 4;
  const int c    = lane & 15;
  const int t0   = blockIdx.x * 16;
  const int kh   = blockIdx.y;
  const int h    = kh * GG + wv;
  float* pb = &Plds[wv * 16 * PST];

  // Q fragments in registers: qa[kk][v] = Q[t0+c][h][kk*4 + 2*hi + v]
  v2f qa[32];
  {
    const float* qrow = qkv + (size_t)(t0 + c) * QKVW + h * DD + 2 * hi;
#pragma unroll
    for (int kk = 0; kk < 32; ++kk) qa[kk] = *(const v2f*)(qrow + kk * 4);
  }

  // online softmax state per D-layout row (row = 8*hi + v); sink = virtual key
  float m_r[8], l_r[8];
  const float sk = sink[h];
#pragma unroll
  for (int v = 0; v < 8; ++v) { m_r[v] = sk; l_r[v] = 1.0f; }
  v8f acc[4];
#pragma unroll
  for (int i = 0; i < 4; ++i) acc[i] = zero8();

  const float scaling = 0.08838834764831845f;  // 128^-0.5
  int s_lo = t0 + 16 - WIN;
  if (s_lo < 0) s_lo = 0;

  for (int s0 = s_lo; s0 < t0 + 16; s0 += 16) {
    __syncthreads();
#if USE_ASYNC_LDS
    // K tile 16x128: 512 b128 async copies, 4/thread; V tile: 2/thread
#pragma unroll
    for (int j = 0; j < 4; ++j) {
      int i = tid + 128 * j;
      int r = i >> 5, c4 = (i & 31) * 4;
      __builtin_amdgcn_global_load_async_to_lds_b128(
          as_global_b128(qkv + (size_t)(s0 + r) * QKVW + QOFF + kh * DD + c4),
          as_lds_b128(&Klds[r * KST + c4]), 0, 0);
    }
#pragma unroll
    for (int j = 0; j < 2; ++j) {
      int i = tid + 128 * j;
      int r = i >> 4, c4 = (i & 15) * 4;
      __builtin_amdgcn_global_load_async_to_lds_b128(
          as_global_b128(qkv + (size_t)(s0 + r) * QKVW + VOFF + kh * VDD + c4),
          as_lds_b128(&Vlds[r * VST + c4]), 0, 0);
    }
    __builtin_amdgcn_s_wait_asynccnt(0);
#else
#pragma unroll
    for (int j = 0; j < 4; ++j) {
      int i = tid + 128 * j;
      int r = i >> 5, c4 = (i & 31) * 4;
      *(f32x4*)(&Klds[r * KST + c4]) =
          *(const f32x4*)(qkv + (size_t)(s0 + r) * QKVW + QOFF + kh * DD + c4);
    }
#pragma unroll
    for (int j = 0; j < 2; ++j) {
      int i = tid + 128 * j;
      int r = i >> 4, c4 = (i & 15) * 4;
      *(f32x4*)(&Vlds[r * VST + c4]) =
          *(const f32x4*)(qkv + (size_t)(s0 + r) * QKVW + VOFF + kh * VDD + c4);
    }
#endif
    __syncthreads();

    // S = Q (16x128) x K^T (128x16): 32 f32 WMMAs
    v8f S = zero8();
#pragma unroll
    for (int kk = 0; kk < 32; ++kk) {
      v2f b = *(const v2f*)(&Klds[c * KST + kk * 4 + 2 * hi]);
      S = __builtin_amdgcn_wmma_f32_16x16x4_f32(
          false, qa[kk], false, b, (short)0, S, false, false);
    }

    // mask + scale + online softmax (row = 8*hi + v, col = s0 + c)
    const int sg = s0 + c;
    float p_e[8];
#pragma unroll
    for (int v = 0; v < 8; ++v) {
      const int tg = t0 + 8 * hi + v;
      const bool ok = (sg <= tg) && (sg + WIN > tg);
      const float val = ok ? S[v] * scaling : -INFINITY;
      float rm = val;
      rm = fmaxf(rm, __shfl_xor(rm, 1));
      rm = fmaxf(rm, __shfl_xor(rm, 2));
      rm = fmaxf(rm, __shfl_xor(rm, 4));
      rm = fmaxf(rm, __shfl_xor(rm, 8));
      const float m_new = fmaxf(m_r[v], rm);
      const float sc = __expf(m_r[v] - m_new);
      const float pe = __expf(val - m_new);  // exp(-inf) -> 0
      float rs = pe;
      rs += __shfl_xor(rs, 1);
      rs += __shfl_xor(rs, 2);
      rs += __shfl_xor(rs, 4);
      rs += __shfl_xor(rs, 8);
      l_r[v] = l_r[v] * sc + rs;
      m_r[v] = m_new;
      p_e[v] = pe;
#pragma unroll
      for (int ch = 0; ch < 4; ++ch) acc[ch][v] *= sc;
    }

    // stage P (D layout) -> LDS -> re-read in A layout
#pragma unroll
    for (int v = 0; v < 8; ++v) pb[(8 * hi + v) * PST + c] = p_e[v];
    __syncthreads();

    // acc += P (16x16) x V (16x64): 16 f32 WMMAs
#pragma unroll
    for (int kk = 0; kk < 4; ++kk) {
      v2f a = *(const v2f*)(&pb[c * PST + kk * 4 + 2 * hi]);
#pragma unroll
      for (int ch = 0; ch < 4; ++ch) {
        v2f b;
        b[0] = Vlds[(kk * 4 + 2 * hi + 0) * VST + ch * 16 + c];
        b[1] = Vlds[(kk * 4 + 2 * hi + 1) * VST + ch * 16 + c];
        acc[ch] = __builtin_amdgcn_wmma_f32_16x16x4_f32(
            false, a, false, b, (short)0, acc[ch], false, false);
      }
    }
  }

  // epilogue: attn_out[t][h*64 + col] = (acc / l) * V_SCALE
#pragma unroll
  for (int v = 0; v < 8; ++v) {
    const int tg = t0 + 8 * hi + v;
    const float wscale = V_SCALE / l_r[v];
    float* op = attn_out + (size_t)tg * OPROJ_K + h * VDD;
#pragma unroll
    for (int ch = 0; ch < 4; ++ch) op[ch * 16 + c] = acc[ch][v] * wscale;
  }
}

// ---------------------------------------------------------------------------
// launch
// ---------------------------------------------------------------------------
extern "C" void kernel_launch(void* const* d_in, const int* in_sizes, int n_in,
                              void* d_out, int out_size, void* d_ws,
                              size_t ws_size, hipStream_t stream) {
  const int*   positions = (const int*)d_in[0];
  const float* hidden    = (const float*)d_in[1];
  const float* Wqkv      = (const float*)d_in[2];
  const float* Wo        = (const float*)d_in[3];
  const float* sink      = (const float*)d_in[4];
  float*       out       = (float*)d_out;

  const int T = in_sizes[0];  // 2048

  float* qkv  = (float*)d_ws;            // T x 5632
  float* attn = qkv + (size_t)T * QKVW;  // T x 2048

  // 1) qkv = hidden @ Wqkv   [T,4096] x [4096,5632]  (bf16x3 WMMA)
  gemm_bf16x3_kernel<<<dim3(T / 128, QKVW / 64), 256, 0, stream>>>(
      hidden, Wqkv, qkv, T, QKVW, HIDDEN_DIM);

  // 2) RoPE(q,k) in place
  rope_kernel<<<dim3(T, HH + KHH), 64, 0, stream>>>(positions, qkv);

  // 3) sliding-window attention with sinks -> attn [T, 2048] (fp32 WMMA)
  attn_kernel<<<dim3(T / 16, KHH), 128, 0, stream>>>(qkv, sink, attn, T);

  // 4) out = attn @ Wo   [T,2048] x [2048,4096]  (bf16x3 WMMA)
  gemm_bf16x3_kernel<<<dim3(T / 128, HIDDEN_DIM / 64), 256, 0, stream>>>(
      attn, Wo, out, T, HIDDEN_DIM, OPROJ_K);
}
